// scaled_dot_product_attention_83769042141351
// MI455X (gfx1250) — matrix operand order repaired and checked by reading, not compiled, not run on hardware
//
#include <hip/hip_runtime.h>
#include <hip/hip_bf16.h>
#include <math.h>

typedef __attribute__((ext_vector_type(16))) __bf16    v16bf;
typedef __attribute__((ext_vector_type(8)))  __bf16    v8bf;
typedef __attribute__((ext_vector_type(8)))  float     v8f;
typedef __attribute__((ext_vector_type(4)))  float     v4f;
typedef __attribute__((ext_vector_type(4)))  unsigned  v4u;

#define NB    2
#define NH    16
#define SQ    2048
#define SKV   2048
#define DH    128
#define QT    64            // q rows per workgroup (4 waves x 16)
#define KT    64            // kv tile
#define NKVT  (SKV / KT)    // kv tiles per (b,h)

// dynamic-LDS partition (bytes)
#define OFS_K  0u           // 2 x 16KB  K tile, bf16 [kv][d]
#define OFS_V  32768u       // 2 x 16KB  V^T tile, bf16 [d][kv]
#define OFS_B  65536u       // 2 x 16KB  bias tile, f32 [kv][q]  (transposed)
#define OFS_P  98304u       // 4 x 2KB   per-wave P strip, bf16 [16][KT]
#define SMEM_BYTES 106496u

static __device__ inline unsigned pk2(float a, float b) {
  union { __bf16 h[2]; unsigned u; } x;
  x.h[0] = (__bf16)a; x.h[1] = (__bf16)b;
  return x.u;
}

// Build a 16-half A fragment from two non-contiguous 16B LDS chunks.
static __device__ inline v16bf ldA(const __bf16* p0, const __bf16* p1) {
  union { struct { v8bf lo; v8bf hi; } s; v16bf v; } u;
  u.s.lo = *(const v8bf*)p0;
  u.s.hi = *(const v8bf*)p1;
  return u.v;
}

// CDNA5 async DMA: 16B global -> LDS, tracked by ASYNCcnt (ISA 15.18.3 op 98).
static __device__ inline void async_g2l_b128(const void* g, unsigned lds_ofs) {
  asm volatile("global_load_async_to_lds_b128 %0, %1, off"
               :: "v"(lds_ofs), "v"((unsigned long long)(size_t)g)
               : "memory");
}
static __device__ inline void wait_async0() {
  asm volatile("s_wait_asynccnt 0x0" ::: "memory");
}
static __device__ inline void wait_ds0() {
  asm volatile("s_wait_dscnt 0x0" ::: "memory");
}

// ---------------- pre-pass 1: flat fp32 -> bf16 (for K) ----------------
__global__ __launch_bounds__(256) void conv_bf16_kernel(
    const float* __restrict__ in, __bf16* __restrict__ out) {
  const int gid = blockIdx.x * 256 + threadIdx.x;   // 8 floats per thread
  const v4f* ip = (const v4f*)in + (size_t)gid * 2;
  v4f a = ip[0], b = ip[1];
  v4u o;
  o.x = pk2(a.x, a.y); o.y = pk2(a.z, a.w);
  o.z = pk2(b.x, b.y); o.w = pk2(b.z, b.w);
  ((v4u*)out)[gid] = o;
}

// ------- pre-pass 2: V -> bf16, transposed, tile-major Vt[bh][kvt][d][KT] -------
__global__ __launch_bounds__(128) void conv_vt_kernel(
    const float* __restrict__ Vg, __bf16* __restrict__ Vtt) {
  __shared__ __bf16 sT[DH * KT];          // [d][kv] for one tile
  const int tid = threadIdx.x;
  const int kvt = blockIdx.x, bh = blockIdx.y;
  {
    const int kvp  = tid & 31;            // kv pair (kv = 2*kvp, 2*kvp+1)
    const int dblk = (tid >> 5) * 32;     // 32 d values
    const float* g0 = Vg + ((size_t)bh * SKV + (size_t)kvt * KT + 2 * kvp) * DH + dblk;
    const float* g1 = g0 + DH;
    unsigned* dp = (unsigned*)sT;         // 32 uints per d-row
#pragma unroll
    for (int i = 0; i < 32; i += 4) {
      v4f a = *(const v4f*)(g0 + i);
      v4f b = *(const v4f*)(g1 + i);
      dp[(dblk + i + 0) * 32 + kvp] = pk2(a.x, b.x);
      dp[(dblk + i + 1) * 32 + kvp] = pk2(a.y, b.y);
      dp[(dblk + i + 2) * 32 + kvp] = pk2(a.z, b.z);
      dp[(dblk + i + 3) * 32 + kvp] = pk2(a.w, b.w);
    }
  }
  __syncthreads();
  {
    const v4u* src = (const v4u*)(sT + tid * KT);
    v4u* dst = (v4u*)(Vtt + (((size_t)bh * NKVT + kvt) * DH + tid) * KT);
#pragma unroll
    for (int i = 0; i < 8; ++i) dst[i] = src[i];
  }
}

// ------- pre-pass 3: bias [q][kv] -> transposed BiasT [kv][q] (tiled) -------
__global__ __launch_bounds__(128) void conv_bt_kernel(
    const float* __restrict__ B, float* __restrict__ Bt) {
  __shared__ float sT[64 * 64];
  const int tid = threadIdx.x;
  const int bx = blockIdx.x;   // kv tile
  const int by = blockIdx.y;   // q tile
#pragma unroll
  for (int i = 0; i < 8; ++i) {
    const int idx4 = tid * 8 + i;          // v4 chunk id 0..1023
    const int row = idx4 >> 4;             // q local
    const int col = (idx4 & 15) * 4;       // kv local
    *(v4f*)(sT + row * 64 + col) =
        *(const v4f*)(B + (size_t)(by * 64 + row) * SKV + bx * 64 + col);
  }
  __syncthreads();
#pragma unroll
  for (int i = 0; i < 8; ++i) {
    const int idx4 = tid * 8 + i;
    const int orow = idx4 >> 4;            // kv local
    const int col  = (idx4 & 15) * 4;      // q local
    v4f v;
    v.x = sT[(col + 0) * 64 + orow];
    v.y = sT[(col + 1) * 64 + orow];
    v.z = sT[(col + 2) * 64 + orow];
    v.w = sT[(col + 3) * 64 + orow];
    *(v4f*)(Bt + (size_t)(bx * 64 + orow) * SQ + by * 64 + col) = v;
  }
}

// ---------------- main flash-attention kernel ----------------
__global__ __launch_bounds__(128) void fa_fwd_kernel(
    const float* __restrict__ Qg, const __bf16* __restrict__ Kbf,
    const __bf16* __restrict__ Vtt, const float* __restrict__ BiasT,
    const int* __restrict__ pCausal, const int* __restrict__ pInfer,
    const int* __restrict__ pAlibi, float* __restrict__ Out) {

  extern __shared__ char smem[];

  const int tid  = threadIdx.x;
  const int wave = tid >> 5;
  const int lane = tid & 31;
  const int l16  = lane & 15;
  const bool hi  = (lane >= 16);

  const int qt   = blockIdx.x;
  const int bh   = blockIdx.y;
  const int head = bh % NH;
  const int q0   = qt * QT;

  const int causal = pCausal[0];
  const int infer  = pInfer[0];
  const int alibi  = pAlibi[0];
  // h=16 is a power of two: slope[h] = (2^(-8/16))^(h+1)
  const float slope  = alibi ? exp2f(-0.5f * (float)(head + 1)) : 0.0f;
  const float sscale = 0.08838834764831845f;   // 1/sqrt(128), folded into Q

  const size_t baseBH = (size_t)bh * SQ * DH;
  const unsigned ldsBase = (unsigned)(size_t)(char*)smem;   // LDS byte offset

  // ---- Q A-fragments directly from global fp32, pre-scaled (once) ----
  v16bf qa[4];
  {
    const float* qrow = Qg + baseBH + (size_t)(q0 + wave * 16 + l16) * DH;
#pragma unroll
    for (int c = 0; c < 4; ++c) {
      const float* p = qrow + c * 32 + (hi ? 8 : 0);
      v4f f0 = *(const v4f*)(p);
      v4f f1 = *(const v4f*)(p + 4);
      v4f f2 = *(const v4f*)(p + 16);
      v4f f3 = *(const v4f*)(p + 20);
      v16bf a;
      a[0]  = (__bf16)(f0.x * sscale); a[1]  = (__bf16)(f0.y * sscale);
      a[2]  = (__bf16)(f0.z * sscale); a[3]  = (__bf16)(f0.w * sscale);
      a[4]  = (__bf16)(f1.x * sscale); a[5]  = (__bf16)(f1.y * sscale);
      a[6]  = (__bf16)(f1.z * sscale); a[7]  = (__bf16)(f1.w * sscale);
      a[8]  = (__bf16)(f2.x * sscale); a[9]  = (__bf16)(f2.y * sscale);
      a[10] = (__bf16)(f2.z * sscale); a[11] = (__bf16)(f2.w * sscale);
      a[12] = (__bf16)(f3.x * sscale); a[13] = (__bf16)(f3.y * sscale);
      a[14] = (__bf16)(f3.z * sscale); a[15] = (__bf16)(f3.w * sscale);
      qa[c] = a;
    }
  }

  // ones B-operand for row-sum-via-WMMA
  v16bf vones;
#pragma unroll
  for (int j = 0; j < 16; ++j) vones[j] = (__bf16)1.0f;

  v8f o[8];
  float m[8], l[8];
#pragma unroll
  for (int n = 0; n < 8; ++n) {
#pragma unroll
    for (int r = 0; r < 8; ++r) o[n][r] = 0.0f;
  }
#pragma unroll
  for (int r = 0; r < 8; ++r) { m[r] = -3.0e38f; l[r] = 0.0f; }

  int kv_end = causal ? (q0 + QT) : SKV;
  if (kv_end > SKV) kv_end = SKV;
  const int nT = kv_end / KT;

  // issue one tile's async DMA batch (K 16KB + V^T 16KB + biasT 16KB)
  auto issueTile = [&](int kvb, int buf) {
    const __bf16* gk = Kbf + ((size_t)bh * SKV + kvb) * DH;
    const __bf16* gv = Vtt + (((size_t)bh * NKVT + (kvb >> 6)) * (size_t)DH) * KT;
    const float*  gb = BiasT + (size_t)kvb * SQ + q0;   // [kv][q]
    const unsigned bk = ldsBase + OFS_K + (unsigned)buf * 16384u;
    const unsigned bv = ldsBase + OFS_V + (unsigned)buf * 16384u;
    const unsigned bb = ldsBase + OFS_B + (unsigned)buf * 16384u;
#pragma unroll
    for (int i = 0; i < 8; ++i) {
      const int ofs = (i * 128 + tid) * 8;      // halfs, 16B chunks
      async_g2l_b128(gk + ofs, bk + (unsigned)ofs * 2u);
      async_g2l_b128(gv + ofs, bv + (unsigned)ofs * 2u);
    }
#pragma unroll
    for (int j = 0; j < 8; ++j) {
      const int c = tid * 8 + j;                // 16B chunk id 0..1023
      const int row = c >> 4, col = (c & 15) * 4;   // kv row, q col
      async_g2l_b128(gb + (size_t)row * SQ + col,
                     bb + (unsigned)(row * 64 + col) * 4u);
    }
  };

  const int qrb = q0 + wave * 16 + (hi ? 8 : 0);   // q row for VGPR idx 0
  int buf = 0;
  issueTile(0, 0);

  for (int it = 0; it < nT; ++it) {
    const int kvb = it * KT;
    wait_async0();     // my DMA batch for tile `it` landed
    wait_ds0();        // my LDS reads of buf^1 fully drained (cross-wave WAR)
    __syncthreads();   // everyone's tile landed / previous buffer free
    if (it + 1 < nT) issueTile(kvb + KT, buf ^ 1);   // overlap with compute

    const __bf16* sK  = (const __bf16*)(smem + OFS_K + buf * 16384u);
    const __bf16* sVt = (const __bf16*)(smem + OFS_V + buf * 16384u);
    const float*  sB  = (const float*)(smem + OFS_B + buf * 16384u);

    // ---- S' = (Q/sqrt(d)) K^T : 4 column tiles x 4 d-chunks ----
    v8f st[4];
#pragma unroll
    for (int t = 0; t < 4; ++t) {
#pragma unroll
      for (int r = 0; r < 8; ++r) st[t][r] = 0.0f;
    }
#pragma unroll
    for (int t = 0; t < 4; ++t) {
      const __bf16* kp = sK + (t * 16 + l16) * DH;
#pragma unroll
      for (int c = 0; c < 4; ++c) {
        v16bf kb = ldA(kp + c * 32 + (hi ? 8 : 0), kp + c * 32 + (hi ? 8 : 0) + 16);
        st[t] = __builtin_amdgcn_wmma_f32_16x16x32_bf16(
            false, qa[c], false, kb, (short)0, st[t], false, false);
      }
    }

    // ---- bias(LDS, [kv][q]: vector loads) + alibi col term (+causal edge) ----
    // Softmax is row-shift invariant: drop -slope*q here, restore in LSE.
    float at[4];
#pragma unroll
    for (int t = 0; t < 4; ++t)
      at[t] = slope * (float)(kvb + t * 16 + l16);

    v4f b0[4], b1[4];
#pragma unroll
    for (int t = 0; t < 4; ++t) {
      const float* bp = sB + (t * 16 + l16) * 64 + wave * 16 + (hi ? 8 : 0);
      b0[t] = *(const v4f*)bp;
      b1[t] = *(const v4f*)(bp + 4);
    }

    const bool needMask = causal && (kvb + KT - 1 > q0 + wave * 16);
    float lm[8];
#pragma unroll
    for (int r = 0; r < 8; ++r) lm[r] = -3.0e38f;

    if (needMask) {
#pragma unroll
      for (int t = 0; t < 4; ++t) {
        const int kvc = kvb + t * 16 + l16;
#pragma unroll
        for (int r = 0; r < 8; ++r) {
          const float bv = (r < 4) ? b0[t][r] : b1[t][r - 4];
          float s = st[t][r] + bv + at[t];
          if (kvc > qrb + r) s = -__builtin_inff();
          st[t][r] = s;
          lm[r] = fmaxf(lm[r], s);
        }
      }
    } else {
#pragma unroll
      for (int t = 0; t < 4; ++t) {
#pragma unroll
        for (int r = 0; r < 8; ++r) {
          const float bv = (r < 4) ? b0[t][r] : b1[t][r - 4];
          float s = st[t][r] + bv + at[t];
          st[t][r] = s;
          lm[r] = fmaxf(lm[r], s);
        }
      }
    }
#pragma unroll
    for (int xm = 8; xm >= 1; xm >>= 1) {
#pragma unroll
      for (int r = 0; r < 8; ++r)
        lm[r] = fmaxf(lm[r], __shfl_xor(lm[r], xm, 32));
    }

    float mn[8], sc[8];
#pragma unroll
    for (int r = 0; r < 8; ++r) {
      mn[r] = fmaxf(m[r], lm[r]);
      sc[r] = __expf(m[r] - mn[r]);
      m[r]  = mn[r];
    }
#pragma unroll
    for (int n = 0; n < 8; ++n) {
#pragma unroll
      for (int r = 0; r < 8; ++r) o[n][r] *= sc[r];
    }

    // ---- P = exp(S'-m'), stage P (bf16) in per-wave LDS ----
    __bf16* pw = (__bf16*)(smem + OFS_P) + wave * (16 * KT);
#pragma unroll
    for (int t = 0; t < 4; ++t) {
#pragma unroll
      for (int r = 0; r < 8; ++r) {
        float p = __expf(st[t][r] - mn[r]);
        pw[((hi ? 8 : 0) + r) * KT + t * 16 + l16] = (__bf16)p;
      }
    }

    // ---- O += P V, and row-sums via ones-WMMA (D = P x 1) ----
    v8f rsv;
#pragma unroll
    for (int r = 0; r < 8; ++r) rsv[r] = 0.0f;
#pragma unroll
    for (int kc = 0; kc < 2; ++kc) {
      const __bf16* pr = pw + l16 * KT + kc * 32 + (hi ? 8 : 0);
      v16bf pa = ldA(pr, pr + 16);
      rsv = __builtin_amdgcn_wmma_f32_16x16x32_bf16(
          false, pa, false, vones, (short)0, rsv, false, false);
#pragma unroll
      for (int n = 0; n < 8; ++n) {
        const __bf16* vp = sVt + (n * 16 + l16) * KT + kc * 32 + (hi ? 8 : 0);
        v16bf vb = ldA(vp, vp + 16);
        o[n] = __builtin_amdgcn_wmma_f32_16x16x32_bf16(
            false, pa, false, vb, (short)0, o[n], false, false);
      }
    }
#pragma unroll
    for (int r = 0; r < 8; ++r) l[r] = l[r] * sc[r] + rsv[r];

    buf ^= 1;
  }

  // ---- epilogue: O / l, stats = m' + log(l) - slope*q (row term restored) ----
  float inv[8], lse[8];
#pragma unroll
  for (int r = 0; r < 8; ++r) {
    inv[r] = 1.0f / l[r];
    lse[r] = m[r] + __logf(l[r]) - slope * (float)(qrb + r);
  }
  float* op = Out + baseBH + (size_t)(q0 + wave * 16) * DH;
#pragma unroll
  for (int n = 0; n < 8; ++n) {
#pragma unroll
    for (int r = 0; r < 8; ++r) {
      const int row = (hi ? 8 : 0) + r;
      op[(size_t)row * DH + n * 16 + l16] = o[n][r] * inv[r];
    }
  }
  if (!infer && l16 == 0) {   // lanes 0 (rows 0-7) and 16 (rows 8-15)
    float* sp = Out + (size_t)NB * NH * SQ * DH +
                (size_t)bh * SQ + q0 + wave * 16 + (hi ? 8 : 0);
#pragma unroll
    for (int r = 0; r < 8; ++r) sp[r] = lse[r];
  }
}

extern "C" void kernel_launch(void* const* d_in, const int* in_sizes, int n_in,
                              void* d_out, int out_size, void* d_ws, size_t ws_size,
                              hipStream_t stream) {
  const float* Qg   = (const float*)d_in[0];
  const float* Kg   = (const float*)d_in[1];
  const float* Vg   = (const float*)d_in[2];
  const float* Bias = (const float*)d_in[3];
  const int* pCausal = (const int*)d_in[4];
  const int* pInfer  = (const int*)d_in[5];
  const int* pAlibi  = (const int*)d_in[6];
  float* out = (float*)d_out;

  // workspace: Kbf (16MB) | Vtt (16MB) | BiasT (16.8MB)
  const size_t nKV = (size_t)NB * NH * SKV * DH;
  __bf16* Kbf = (__bf16*)d_ws;
  __bf16* Vtt = (__bf16*)((char*)d_ws + nKV * sizeof(__bf16));
  float*  Bt  = (float*)((char*)d_ws + 2 * nKV * sizeof(__bf16));

  conv_bf16_kernel<<<dim3((unsigned)(nKV / (256 * 8))), dim3(256), 0, stream>>>(Kg, Kbf);
  conv_vt_kernel<<<dim3(NKVT, NB * NH), dim3(128), 0, stream>>>(Vg, Vtt);
  conv_bt_kernel<<<dim3(SKV / 64, SQ / 64), dim3(128), 0, stream>>>(Bias, Bt);

  dim3 grid(SQ / QT, NB * NH);
  fa_fwd_kernel<<<grid, dim3(128), SMEM_BYTES, stream>>>(Qg, Kbf, Vtt, Bt,
                                                         pCausal, pInfer, pAlibi, out);
}
